// MultiHeadedAttention_51522427683067
// MI455X (gfx1250) — matrix operand order repaired and checked by reading, not compiled, not run on hardware
//
#include <hip/hip_runtime.h>
#include <hip/hip_bf16.h>

// MI455X / gfx1250, wave32. WMMA 16x16x32 f16 -> f32 throughout.

typedef _Float16 h16 __attribute__((ext_vector_type(16)));
typedef _Float16 h8  __attribute__((ext_vector_type(8)));
typedef float    f8  __attribute__((ext_vector_type(8)));
typedef float    f4v __attribute__((ext_vector_type(4)));

#define HIDDEN 1024
#define HEADS  16
#define HEAD   64
#define SEQ    1024   // T == N == 1024
#define BATCH  4

// ---------------------------------------------------------------------------
// Kernel 1: Y = X @ W^T + b for Q/K/V.
//   Q, K stored f16 as [b, h, pos, 64]; V stored f16 TRANSPOSED as [b, h, 64, pos]
//   so the attention PV gemm loads V B-fragments contiguously.
// Block = 256 threads (8 waves) computes a 128x128 output tile.
//   - W k-slice (128 cols x 32 k) staged f32->f16 in LDS once per block (8 KB).
//   - Each wave owns 16 rows; its A-fragment is reused across 8 n-tiles
//     (8 WMMAs per A load). B-fragments preloaded as a batch so the compiler
//     can overlap ds_load_b128s with WMMAs instead of draining dscnt each time.
// grid = (8 n-groups, 32 m-groups, 3 matrices).
// ---------------------------------------------------------------------------
__global__ void qkv_proj_kernel(const float* __restrict__ x,
                                const float* __restrict__ enc,
                                const float* __restrict__ Wq, const float* __restrict__ bq,
                                const float* __restrict__ Wk, const float* __restrict__ bk,
                                const float* __restrict__ Wv, const float* __restrict__ bv,
                                _Float16* __restrict__ Qws,
                                _Float16* __restrict__ Kws,
                                _Float16* __restrict__ Vws) {
    __shared__ _Float16 Wtile[128][32];      // 8 KB

    const int tid  = threadIdx.x;
    const int lane = tid & 31;
    const int wid  = tid >> 5;               // 0..7
    const int half = (lane >> 4) & 1;
    const int ln   = lane & 15;

    const int nbase = blockIdx.x * 128;      // output col base
    const int mbase = blockIdx.y * 128;      // output row base
    const int sel   = blockIdx.z;            // 0=Q 1=K 2=V

    const float* A    = (sel == 0) ? x  : enc;
    const float* W    = (sel == 0) ? Wq : ((sel == 1) ? Wk : Wv);
    const float* bias = (sel == 0) ? bq : ((sel == 1) ? bk : bv);
    _Float16*    dst  = (sel == 0) ? Qws : ((sel == 1) ? Kws : Vws);

    // staging assignment: thread -> (W row, 16-wide k group)
    const int srow = tid >> 1;               // 0..127
    const int skg  = (tid & 1) * 16;         // 0 or 16

    const float* arow = A + (size_t)(mbase + wid * 16 + ln) * HIDDEN;
    const float* wrow = W + (size_t)(nbase + srow) * HIDDEN;

    f8 acc[8] = {{}, {}, {}, {}, {}, {}, {}, {}};

    for (int k0 = 0; k0 < HIDDEN; k0 += 32) {
        // ---- cooperative stage: W[128][k0..k0+31] f32 -> f16 in LDS
        {
            f4v w0 = *(const f4v*)(wrow + k0 + skg);
            f4v w1 = *(const f4v*)(wrow + k0 + skg + 4);
            f4v w2 = *(const f4v*)(wrow + k0 + skg + 8);
            f4v w3 = *(const f4v*)(wrow + k0 + skg + 12);
            h16 hv;
#pragma unroll
            for (int e = 0; e < 4; ++e) {
                hv[e]      = (_Float16)w0[e];
                hv[4 + e]  = (_Float16)w1[e];
                hv[8 + e]  = (_Float16)w2[e];
                hv[12 + e] = (_Float16)w3[e];
            }
            *(h16*)(&Wtile[srow][skg]) = hv;
        }

        // ---- per-wave A fragment (16 rows x 32 k), f32 -> f16
        h16 afrag;
        {
            const int ab = k0 + half * 16;
            f4v a0 = *(const f4v*)(arow + ab);
            f4v a1 = *(const f4v*)(arow + ab + 4);
            f4v a2 = *(const f4v*)(arow + ab + 8);
            f4v a3 = *(const f4v*)(arow + ab + 12);
#pragma unroll
            for (int e = 0; e < 4; ++e) {
                afrag[e]      = (_Float16)a0[e];
                afrag[4 + e]  = (_Float16)a1[e];
                afrag[8 + e]  = (_Float16)a2[e];
                afrag[12 + e] = (_Float16)a3[e];
            }
        }

        __syncthreads();   // staging visible to all waves

        // ---- preload all 8 B fragments, then batch the 8 WMMAs
        h16 bfr[8];
#pragma unroll
        for (int nt = 0; nt < 8; ++nt)
            bfr[nt] = *(const h16*)(&Wtile[nt * 16 + ln][half * 16]);
#pragma unroll
        for (int nt = 0; nt < 8; ++nt)
            acc[nt] = __builtin_amdgcn_wmma_f32_16x16x32_f16(
                          false, afrag, false, bfr[nt], (short)0, acc[nt], false, false);

        __syncthreads();   // protect LDS before next staging
    }

    // ---- epilogue: bias + store f16 (V transposed)
#pragma unroll
    for (int nt = 0; nt < 8; ++nt) {
        const int gcol = nbase + nt * 16 + ln;
        const float bval = bias[gcol];
        const int h = gcol >> 6;
        const int d = gcol & 63;
#pragma unroll
        for (int r = 0; r < 8; ++r) {
            const int gm = mbase + wid * 16 + r + half * 8;
            const int b  = gm >> 10;
            const int t  = gm & 1023;
            const _Float16 val = (_Float16)(acc[nt][r] + bval);
            if (sel == 2) {   // V: [b, h, d, pos]
                dst[(((size_t)(b * HEADS + h)) * HEAD + d) * SEQ + t] = val;
            } else {          // Q, K: [b, h, pos, d]
                dst[(((size_t)(b * HEADS + h)) * SEQ + t) * HEAD + d] = val;
            }
        }
    }
}

// ---------------------------------------------------------------------------
// Kernel 2: causal attention, flash style. One wave per (b*h, 16-query tile).
// grid = (64 bh, 8 t-groups), block = 256 (8 waves).
// ---------------------------------------------------------------------------
__global__ void attn_kernel(const _Float16* __restrict__ Qws,
                            const _Float16* __restrict__ Kws,
                            const _Float16* __restrict__ Vt,   // [b,h,64,pos]
                            float* __restrict__ out) {
    __shared__ _Float16 pbuf[8][16 * 32];    // per-wave 16x32 P tile (f16)

    const int lane = threadIdx.x & 31;
    const int wid  = threadIdx.x >> 5;
    const int half = (lane >> 4) & 1;
    const int ln   = lane & 15;

    const int bh = blockIdx.x;               // 0..63
    const int b  = bh >> 4;
    const int h  = bh & 15;
    const int t_tile = blockIdx.y * 8 + wid; // 0..63
    const int t0 = t_tile * 16;

    const _Float16* Qb = Qws + (size_t)bh * SEQ * HEAD;
    const _Float16* Kb = Kws + (size_t)bh * SEQ * HEAD;
    const _Float16* Vb = Vt  + (size_t)bh * HEAD * SEQ;

    // Q as two A-fragments covering head-dim 0..31 and 32..63
    h16 qa[2];
    {
        const _Float16* qrow = Qb + (size_t)(t0 + ln) * HEAD;
#pragma unroll
        for (int f = 0; f < 2; ++f) {
            const int kb = f * 32 + half * 16;
            h8 lo = *(const h8*)(qrow + kb);
            h8 hi = *(const h8*)(qrow + kb + 8);
#pragma unroll
            for (int e = 0; e < 8; ++e) { qa[f][e] = lo[e]; qa[f][8 + e] = hi[e]; }
        }
    }

    f8 o0 = {}, o1 = {}, o2 = {}, o3 = {};
    float m_[8], l_[8];
#pragma unroll
    for (int r = 0; r < 8; ++r) { m_[r] = -1e30f; l_[r] = 0.0f; }

    const int nlimit = t0 + 16;              // exclusive kv bound (causal)
    for (int n0 = 0; n0 < nlimit; n0 += 32) {
        // prefetch next chunk's K rows and V^T rows (speculative, L2-friendly)
        __builtin_prefetch(Kb + (size_t)(n0 + 32 + ln) * HEAD, 0, 0);
        __builtin_prefetch(Vb + (size_t)(ln) * SEQ + n0 + 32, 0, 0);

        // ---- preload all K^T fragments, then batch the 4 S WMMAs
        h16 kf[4];   // [f=0: d0..31][f=1: d32..63] x [tile0, tile1]
#pragma unroll
        for (int f = 0; f < 2; ++f) {
            const int d0 = f * 32;
            kf[f * 2 + 0] = *(const h16*)(Kb + (size_t)(n0 + ln)      * HEAD + d0 + half * 16);
            kf[f * 2 + 1] = *(const h16*)(Kb + (size_t)(n0 + 16 + ln) * HEAD + d0 + half * 16);
        }
        f8 s0 = {}, s1 = {};
        s0 = __builtin_amdgcn_wmma_f32_16x16x32_f16(false, qa[0], false, kf[0], (short)0, s0, false, false);
        s1 = __builtin_amdgcn_wmma_f32_16x16x32_f16(false, qa[0], false, kf[1], (short)0, s1, false, false);
        s0 = __builtin_amdgcn_wmma_f32_16x16x32_f16(false, qa[1], false, kf[2], (short)0, s0, false, false);
        s1 = __builtin_amdgcn_wmma_f32_16x16x32_f16(false, qa[1], false, kf[3], (short)0, s1, false, false);

        // ---- scale, causal mask, online softmax (row = 16-lane half reduce)
#pragma unroll
        for (int r = 0; r < 8; ++r) {
            const int row = t0 + r + half * 8;
            const int c0  = n0 + ln;
            const int c1  = n0 + 16 + ln;
            float v0 = s0[r] * 0.125f;          // 1/sqrt(64)
            float v1 = s1[r] * 0.125f;
            if (c0 > row) v0 = -1e30f;
            if (c1 > row) v1 = -1e30f;

            float cm = fmaxf(v0, v1);
            cm = fmaxf(cm, __shfl_xor(cm, 1));
            cm = fmaxf(cm, __shfl_xor(cm, 2));
            cm = fmaxf(cm, __shfl_xor(cm, 4));
            cm = fmaxf(cm, __shfl_xor(cm, 8));

            const float nm = fmaxf(m_[r], cm);
            const float sc = __expf(m_[r] - nm);
            const float p0 = __expf(v0 - nm);
            const float p1 = __expf(v1 - nm);
            float rs = p0 + p1;
            rs += __shfl_xor(rs, 1);
            rs += __shfl_xor(rs, 2);
            rs += __shfl_xor(rs, 4);
            rs += __shfl_xor(rs, 8);

            l_[r] = l_[r] * sc + rs;
            m_[r] = nm;
            o0[r] *= sc; o1[r] *= sc; o2[r] *= sc; o3[r] *= sc;

            // spill P (C layout) to LDS as f16
            pbuf[wid][(r + half * 8) * 32 + ln]      = (_Float16)p0;
            pbuf[wid][(r + half * 8) * 32 + 16 + ln] = (_Float16)p1;
        }

        // ---- reload P as an A-fragment: k-mapping = two contiguous 8-half
        // runs -> 2x ds_load_b128. DS ops are in-order within a wave.
        h16 pa;
        {
            const _Float16* pr = &pbuf[wid][ln * 32 + half * 16];
            h8 lo = *(const h8*)(pr);
            h8 hi = *(const h8*)(pr + 8);
#pragma unroll
            for (int e = 0; e < 8; ++e) { pa[e] = lo[e]; pa[8 + e] = hi[e]; }
        }

        // ---- preload all 4 V^T fragments, then batch the 4 PV WMMAs
        h16 vf[4];
#pragma unroll
        for (int dt = 0; dt < 4; ++dt)
            vf[dt] = *(const h16*)(Vb + (size_t)(dt * 16 + ln) * SEQ + n0 + half * 16);
        o0 = __builtin_amdgcn_wmma_f32_16x16x32_f16(false, pa, false, vf[0], (short)0, o0, false, false);
        o1 = __builtin_amdgcn_wmma_f32_16x16x32_f16(false, pa, false, vf[1], (short)0, o1, false, false);
        o2 = __builtin_amdgcn_wmma_f32_16x16x32_f16(false, pa, false, vf[2], (short)0, o2, false, false);
        o3 = __builtin_amdgcn_wmma_f32_16x16x32_f16(false, pa, false, vf[3], (short)0, o3, false, false);
    }

    // ---- normalize and store: out[b, t, h*64 + dt*16 + ln], f32
#pragma unroll
    for (int r = 0; r < 8; ++r) {
        const float inv = 1.0f / l_[r];
        const int t = t0 + r + half * 8;
        float* orow = out + ((size_t)b * SEQ + t) * HIDDEN + h * HEAD;
        orow[0 * 16 + ln] = o0[r] * inv;
        orow[1 * 16 + ln] = o1[r] * inv;
        orow[2 * 16 + ln] = o2[r] * inv;
        orow[3 * 16 + ln] = o3[r] * inv;
    }
}

extern "C" void kernel_launch(void* const* d_in, const int* in_sizes, int n_in,
                              void* d_out, int out_size, void* d_ws, size_t ws_size,
                              hipStream_t stream) {
    const float* x   = (const float*)d_in[0];
    const float* enc = (const float*)d_in[1];
    const float* Wq  = (const float*)d_in[2];
    const float* bq  = (const float*)d_in[3];
    const float* Wk  = (const float*)d_in[4];
    const float* bk  = (const float*)d_in[5];
    const float* Wv  = (const float*)d_in[6];
    const float* bv  = (const float*)d_in[7];
    float* out = (float*)d_out;

    const size_t per = (size_t)BATCH * HEADS * SEQ * HEAD;  // 4 Mi f16 = 8 MiB
    _Float16* Qws = (_Float16*)d_ws;
    _Float16* Kws = Qws + per;
    _Float16* Vws = Kws + per;       // V stored transposed [b,h,64,pos]

    qkv_proj_kernel<<<dim3(8, 32, 3), 256, 0, stream>>>(
        x, enc, Wq, bq, Wk, bk, Wv, bv, Qws, Kws, Vws);

    attn_kernel<<<dim3(64, 8), 256, 0, stream>>>(Qws, Kws, Vws, out);
}
